// ProteinEncoder_90245852823574
// MI455X (gfx1250) — matrix operand order repaired and checked by reading, not compiled, not run on hardware
//
#include <hip/hip_runtime.h>

#define N_NODES 20000
#define N_EDGES 640000
#define NF 128
#define EF 32
#define HD 128
#define NL 3

typedef __attribute__((ext_vector_type(16))) _Float16 v16h;
typedef __attribute__((ext_vector_type(8)))  float    v8f;
typedef __attribute__((ext_vector_type(4)))  unsigned int u32x4;

// LDS halfword budgets
#define MSG_W_HALVES (288 * 128 + 128 * 128)          // 53248 f16
#define MSG_LDS_BYTES ((MSG_W_HALVES + 8 * 16 * 128) * 2)   // 139264 B
#define UPD_W_HALVES (256 * 128 + 128 * 128)          // 49152 f16
#define UPD_LDS_BYTES ((UPD_W_HALVES + 8 * 16 * 128) * 2)   // 131072 B

union HFrag {
    v16h v;
    _Float16 h[16];
    u32x4 q[2];
};

// ---- WMMA helpers -------------------------------------------------------

__device__ __forceinline__ v8f wmma_f16(v16h a, v16h b, v8f c) {
    return __builtin_amdgcn_wmma_f32_16x16x32_f16(
        /*neg_a=*/false, a, /*neg_b=*/false, b,
        /*c_mod=*/(short)0, c, /*reuse_a=*/false, /*reuse_b=*/false);
}

// A-operand (16x32 f16): lane row r = lane%16; two contiguous 8-elem K
// segments at K = half*8 and K = 16 + half*8.
__device__ __forceinline__ v16h load_a_frag(const _Float16* row, int half) {
    HFrag f;
    f.q[0] = *(const u32x4*)(row + half * 8);
    f.q[1] = *(const u32x4*)(row + 16 + half * 8);
    return f.v;
}

__device__ __forceinline__ v16h load_a_frag_f32(const float* row, int half) {
    HFrag f;
#pragma unroll
    for (int i = 0; i < 8; ++i) f.h[i] = (_Float16)row[half * 8 + i];
#pragma unroll
    for (int i = 0; i < 8; ++i) f.h[8 + i] = (_Float16)row[16 + half * 8 + i];
    return f.v;
}

// B-operand from fragment-packed LDS: lane's 16 f16 are contiguous (32 B).
__device__ __forceinline__ v16h load_b_frag(const _Float16* p) {
    HFrag f;
    f.q[0] = ((const u32x4*)p)[0];
    f.q[1] = ((const u32x4*)p)[1];
    return f.v;
}

__device__ __forceinline__ void zero_acc(v8f* acc) {
#pragma unroll
    for (int j = 0; j < 8; ++j)
#pragma unroll
        for (int i = 0; i < 8; ++i) acc[j][i] = 0.0f;
}

// Launder a value through a VGPR so LICM cannot hoist address computations
// (prevents the compiler from pre-materializing every B fragment and
// spilling them to scratch, as observed in round 1).
__device__ __forceinline__ unsigned launder_u32(unsigned x) {
    asm volatile("" : "+v"(x));
    return x;
}

// ---- Weight pre-pack: KxN (N=128) f32 row-major -> B-fragment-major f16 ----
// packed[(((c*8 + j)*32 + lane)*16 + s] = W[c*32 + (s/8)*16 + half*8 + s%8][j*16 + lane%16]
__global__ void pack_w_kernel(const float* __restrict__ W, _Float16* __restrict__ P, int K) {
    int idx = blockIdx.x * blockDim.x + threadIdx.x;
    int total = K * 128;
    if (idx >= total) return;
    int s = idx & 15;
    int rest = idx >> 4;
    int lane = rest & 31; rest >>= 5;
    int j = rest & 7;
    int c = rest >> 3;
    int half = lane >> 4, nl = lane & 15;
    int k = c * 32 + ((s >> 3) << 4) + half * 8 + (s & 7);
    int n = j * 16 + nl;
    P[idx] = (_Float16)W[k * 128 + n];
}

// ---- Embedding kernels --------------------------------------------------

__global__ void node_embed_kernel(const float* __restrict__ coords,
                                  const float* __restrict__ W, const float* __restrict__ b,
                                  float* __restrict__ h, _Float16* __restrict__ h16) {
    int idx = blockIdx.x * blockDim.x + threadIdx.x;
    if (idx >= N_NODES * NF) return;
    int f = idx & (NF - 1);
    int node = idx >> 7;
    const float* row = coords + (size_t)node * 6;
    float s = b[f];
#pragma unroll
    for (int k = 0; k < 6; ++k) s += row[k] * W[k * NF + f];
    h[idx] = s;
    h16[idx] = (_Float16)s;
}

__global__ void edge_embed_kernel(const float* __restrict__ dist,
                                  const float* __restrict__ W, const float* __restrict__ b,
                                  _Float16* __restrict__ e16) {
    int idx = blockIdx.x * blockDim.x + threadIdx.x;
    if (idx >= N_EDGES * EF) return;
    int f = idx & (EF - 1);
    int e = idx >> 5;
    const float* row = dist + (size_t)e * EF;
    float s = b[f];
#pragma unroll
    for (int k = 0; k < EF; ++k) s += row[k] * W[k * EF + f];
    e16[idx] = (_Float16)s;
}

__global__ void zero_kernel(float* __restrict__ p, int n) {
    int idx = blockIdx.x * blockDim.x + threadIdx.x;
    if (idx < n) p[idx] = 0.0f;
}

// ---- Edge message MLP + scatter-add ------------------------------------
// One wave = 16 edges. concat(h[src],h[dst],e) [16,288] -> relu GEMM -> GEMM
// -> atomic scatter into agg[dst]. Weights live in LDS.
__global__ void __launch_bounds__(256) msg_kernel(
    const _Float16* __restrict__ h16, const _Float16* __restrict__ e16,
    const int* __restrict__ src_idx, const int* __restrict__ dst_idx,
    const _Float16* __restrict__ W1p, const float* __restrict__ b1,
    const _Float16* __restrict__ W2p, const float* __restrict__ b2,
    float* __restrict__ agg, int n_edges) {
    extern __shared__ _Float16 lds[];
    // [0, 36864): W1 packed; [36864, 53248): W2 packed; then 8 x 2048 tbuf
    {
        u32x4* dl = (u32x4*)lds;
        const u32x4* g1 = (const u32x4*)W1p;
        for (int i = threadIdx.x; i < (288 * 128) / 8; i += 256) dl[i] = g1[i];
        const u32x4* g2 = (const u32x4*)W2p;
        for (int i = threadIdx.x; i < (128 * 128) / 8; i += 256) dl[(288 * 128) / 8 + i] = g2[i];
    }
    __syncthreads();

    const int lane = threadIdx.x & 31;
    const int wid = threadIdx.x >> 5;
    const int half = lane >> 4;
    const int nl = lane & 15;
    _Float16* tbuf = lds + MSG_W_HALVES + wid * (16 * 128);
    const int gwave = blockIdx.x * 8 + wid;
    const int nwaves = gridDim.x * 8;
    const int ntiles = n_edges >> 4;

    float bb1[8], bb2[8];
#pragma unroll
    for (int j = 0; j < 8; ++j) { bb1[j] = b1[j * 16 + nl]; bb2[j] = b2[j * 16 + nl]; }

    for (int tile = gwave; tile < ntiles; tile += nwaves) {
        // per-iteration laundered LDS bases: keep B-frag ds loads inside loop
        const _Float16* W1s = lds + launder_u32(0);
        const _Float16* W2s = lds + 288 * 128 + launder_u32(0);

        const int eb = tile * 16;
        const int sm = src_idx[eb + nl];
        const int dm = dst_idx[eb + nl];

        v8f acc[8];
        zero_acc(acc);
        // --- GEMM1: [16,288] x [288,128], K chunks of 32 (h[src]|h[dst]|e) ---
#pragma unroll
        for (int c = 0; c < 9; ++c) {
            const _Float16* arow;
            if (c < 4)      arow = h16 + (size_t)sm * NF + c * 32;
            else if (c < 8) arow = h16 + (size_t)dm * NF + (c - 4) * 32;
            else            arow = e16 + (size_t)(eb + nl) * EF;
            v16h a = load_a_frag(arow, half);
#pragma unroll
            for (int j = 0; j < 8; ++j) {
                v16h b = load_b_frag(W1s + ((c * 8 + j) * 32 + lane) * 16);
                acc[j] = wmma_f16(a, b, acc[j]);
            }
        }
        // bias + relu, stage transposed through LDS for second GEMM
#pragma unroll
        for (int j = 0; j < 8; ++j) {
#pragma unroll
            for (int v = 0; v < 8; ++v) {
                float t = acc[j][v] + bb1[j];
                t = t > 0.0f ? t : 0.0f;
                tbuf[(v + 8 * half) * 128 + j * 16 + nl] = (_Float16)t;
            }
        }
        asm volatile("s_wait_dscnt 0x0" ::: "memory");

        v8f acc2[8];
        zero_acc(acc2);
        // --- GEMM2: [16,128] x [128,128] ---
#pragma unroll
        for (int c = 0; c < 4; ++c) {
            v16h a = load_a_frag(tbuf + nl * 128 + c * 32, half);
#pragma unroll
            for (int j = 0; j < 8; ++j) {
                v16h b = load_b_frag(W2s + ((c * 8 + j) * 32 + lane) * 16);
                acc2[j] = wmma_f16(a, b, acc2[j]);
            }
        }
        // scatter-add rows to agg[dst]
#pragma unroll
        for (int v = 0; v < 8; ++v) {
            int d = __shfl(dm, v + 8 * half, 32);
#pragma unroll
            for (int j = 0; j < 8; ++j) {
                unsafeAtomicAdd(&agg[(size_t)d * HD + j * 16 + nl], acc2[j][v] + bb2[j]);
            }
        }
    }
}

// ---- Node update MLP: concat(h, agg) -> relu GEMM -> GEMM -> +h -> tmp ----
__global__ void __launch_bounds__(256) upd_kernel(
    const _Float16* __restrict__ h16, const float* __restrict__ hfp,
    const float* __restrict__ agg,
    const _Float16* __restrict__ W1p, const float* __restrict__ b1,
    const _Float16* __restrict__ W2p, const float* __restrict__ b2,
    float* __restrict__ tmp, int n_nodes) {
    extern __shared__ _Float16 lds[];
    {
        u32x4* dl = (u32x4*)lds;
        const u32x4* g1 = (const u32x4*)W1p;
        for (int i = threadIdx.x; i < (256 * 128) / 8; i += 256) dl[i] = g1[i];
        const u32x4* g2 = (const u32x4*)W2p;
        for (int i = threadIdx.x; i < (128 * 128) / 8; i += 256) dl[(256 * 128) / 8 + i] = g2[i];
    }
    __syncthreads();

    const int lane = threadIdx.x & 31;
    const int wid = threadIdx.x >> 5;
    const int half = lane >> 4;
    const int nl = lane & 15;
    _Float16* tbuf = lds + UPD_W_HALVES + wid * (16 * 128);
    const int gwave = blockIdx.x * 8 + wid;
    const int nwaves = gridDim.x * 8;
    const int ntiles = n_nodes >> 4;

    float bb1[8], bb2[8];
#pragma unroll
    for (int j = 0; j < 8; ++j) { bb1[j] = b1[j * 16 + nl]; bb2[j] = b2[j * 16 + nl]; }

    for (int tile = gwave; tile < ntiles; tile += nwaves) {
        const _Float16* W1s = lds + launder_u32(0);
        const _Float16* W2s = lds + 256 * 128 + launder_u32(0);

        const int nb = tile * 16;
        const int node = nb + nl;

        v8f acc[8];
        zero_acc(acc);
#pragma unroll
        for (int c = 0; c < 8; ++c) {
            v16h a;
            if (c < 4) a = load_a_frag(h16 + (size_t)node * NF + c * 32, half);
            else       a = load_a_frag_f32(agg + (size_t)node * HD + (c - 4) * 32, half);
#pragma unroll
            for (int j = 0; j < 8; ++j) {
                v16h b = load_b_frag(W1s + ((c * 8 + j) * 32 + lane) * 16);
                acc[j] = wmma_f16(a, b, acc[j]);
            }
        }
#pragma unroll
        for (int j = 0; j < 8; ++j) {
#pragma unroll
            for (int v = 0; v < 8; ++v) {
                float t = acc[j][v] + bb1[j];
                t = t > 0.0f ? t : 0.0f;
                tbuf[(v + 8 * half) * 128 + j * 16 + nl] = (_Float16)t;
            }
        }
        asm volatile("s_wait_dscnt 0x0" ::: "memory");

        v8f acc2[8];
        zero_acc(acc2);
#pragma unroll
        for (int c = 0; c < 4; ++c) {
            v16h a = load_a_frag(tbuf + nl * 128 + c * 32, half);
#pragma unroll
            for (int j = 0; j < 8; ++j) {
                v16h b = load_b_frag(W2s + ((c * 8 + j) * 32 + lane) * 16);
                acc2[j] = wmma_f16(a, b, acc2[j]);
            }
        }
        // residual: tmp = h + delta (pre-LayerNorm)
#pragma unroll
        for (int v = 0; v < 8; ++v) {
            int row = nb + v + 8 * half;
#pragma unroll
            for (int j = 0; j < 8; ++j) {
                int n = j * 16 + nl;
                tmp[(size_t)row * NF + n] = hfp[(size_t)row * NF + n] + acc2[j][v] + bb2[j];
            }
        }
    }
}

// ---- LayerNorm: one wave per node (128 features) ------------------------
__global__ void ln_kernel(const float* __restrict__ tmp,
                          const float* __restrict__ g, const float* __restrict__ b,
                          float* __restrict__ h, _Float16* __restrict__ h16, int n_nodes) {
    int wid = (blockIdx.x * blockDim.x + threadIdx.x) >> 5;
    int lane = threadIdx.x & 31;
    if (wid >= n_nodes) return;
    const float* row = tmp + (size_t)wid * NF;
    float x[4];
    float s = 0.0f;
#pragma unroll
    for (int i = 0; i < 4; ++i) { x[i] = row[lane + 32 * i]; s += x[i]; }
#pragma unroll
    for (int off = 16; off >= 1; off >>= 1) s += __shfl_xor(s, off, 32);
    float mu = s * (1.0f / 128.0f);
    float vs = 0.0f;
#pragma unroll
    for (int i = 0; i < 4; ++i) { float d = x[i] - mu; vs += d * d; }
#pragma unroll
    for (int off = 16; off >= 1; off >>= 1) vs += __shfl_xor(vs, off, 32);
    float rstd = rsqrtf(vs * (1.0f / 128.0f) + 1e-5f);
#pragma unroll
    for (int i = 0; i < 4; ++i) {
        int n = lane + 32 * i;
        float y = (x[i] - mu) * rstd * g[n] + b[n];
        h[(size_t)wid * NF + n] = y;
        h16[(size_t)wid * NF + n] = (_Float16)y;
    }
}

// ---- Output projection: out = h @ W_out + b_out -------------------------
__global__ void __launch_bounds__(256) out_kernel(
    const _Float16* __restrict__ h16, const _Float16* __restrict__ Wp,
    const float* __restrict__ bo, float* __restrict__ out, int n_nodes) {
    __shared__ _Float16 wlds[128 * 128];
    {
        u32x4* dl = (u32x4*)wlds;
        const u32x4* gw = (const u32x4*)Wp;
        for (int i = threadIdx.x; i < (128 * 128) / 8; i += 256) dl[i] = gw[i];
    }
    __syncthreads();

    const int lane = threadIdx.x & 31;
    const int wid = threadIdx.x >> 5;
    const int half = lane >> 4;
    const int nl = lane & 15;
    const int gwave = blockIdx.x * 8 + wid;
    const int nwaves = gridDim.x * 8;
    const int ntiles = n_nodes >> 4;

    float bb[8];
#pragma unroll
    for (int j = 0; j < 8; ++j) bb[j] = bo[j * 16 + nl];

    for (int tile = gwave; tile < ntiles; tile += nwaves) {
        const _Float16* Ws = wlds + launder_u32(0);
        const int nb = tile * 16;
        v8f acc[8];
        zero_acc(acc);
#pragma unroll
        for (int c = 0; c < 4; ++c) {
            v16h a = load_a_frag(h16 + (size_t)(nb + nl) * NF + c * 32, half);
#pragma unroll
            for (int j = 0; j < 8; ++j) {
                v16h b = load_b_frag(Ws + ((c * 8 + j) * 32 + lane) * 16);
                acc[j] = wmma_f16(a, b, acc[j]);
            }
        }
#pragma unroll
        for (int v = 0; v < 8; ++v) {
            int row = nb + v + 8 * half;
#pragma unroll
            for (int j = 0; j < 8; ++j) {
                out[(size_t)row * HD + j * 16 + nl] = acc[j][v] + bb[j];
            }
        }
    }
}

// ---- Launch -------------------------------------------------------------

extern "C" void kernel_launch(void* const* d_in, const int* in_sizes, int n_in,
                              void* d_out, int out_size, void* d_ws, size_t ws_size,
                              hipStream_t stream) {
    const float* node_coords = (const float*)d_in[0];
    const int*   edge_index  = (const int*)d_in[1];
    const float* edge_dist   = (const float*)d_in[2];
    const float* W_node = (const float*)d_in[3];
    const float* b_node = (const float*)d_in[4];
    const float* W_edge = (const float*)d_in[5];
    const float* b_edge = (const float*)d_in[6];
    const float* msg_W1 = (const float*)d_in[7];
    const float* msg_b1 = (const float*)d_in[8];
    const float* msg_W2 = (const float*)d_in[9];
    const float* msg_b2 = (const float*)d_in[10];
    const float* upd_W1 = (const float*)d_in[11];
    const float* upd_b1 = (const float*)d_in[12];
    const float* upd_W2 = (const float*)d_in[13];
    const float* upd_b2 = (const float*)d_in[14];
    const float* ln_g   = (const float*)d_in[15];
    const float* ln_b   = (const float*)d_in[16];
    const float* W_out  = (const float*)d_in[17];
    const float* b_out  = (const float*)d_in[18];

    char* ws = (char*)d_ws;
    size_t off = 0;
    auto alloc = [&](size_t bytes) -> void* {
        void* p = ws + off;
        off += (bytes + 255) & ~(size_t)255;
        return p;
    };
    float*    h    = (float*)alloc((size_t)N_NODES * NF * 4);
    float*    tmp  = (float*)alloc((size_t)N_NODES * NF * 4);
    float*    agg  = (float*)alloc((size_t)N_NODES * HD * 4);
    _Float16* h16  = (_Float16*)alloc((size_t)N_NODES * NF * 2);
    _Float16* e16  = (_Float16*)alloc((size_t)N_EDGES * EF * 2);
    _Float16* msgW1p = (_Float16*)alloc((size_t)NL * 288 * 128 * 2);
    _Float16* msgW2p = (_Float16*)alloc((size_t)NL * 128 * 128 * 2);
    _Float16* updW1p = (_Float16*)alloc((size_t)NL * 256 * 128 * 2);
    _Float16* updW2p = (_Float16*)alloc((size_t)NL * 128 * 128 * 2);
    _Float16* Woutp  = (_Float16*)alloc((size_t)128 * 128 * 2);

    // allow >64KB dynamic LDS on the two big kernels (WGP has 320KB)
    hipFuncSetAttribute((const void*)msg_kernel,
                        hipFuncAttributeMaxDynamicSharedMemorySize, MSG_LDS_BYTES);
    hipFuncSetAttribute((const void*)upd_kernel,
                        hipFuncAttributeMaxDynamicSharedMemorySize, UPD_LDS_BYTES);

    dim3 blk(256);

    // Pack weights into B-fragment-major f16
    for (int l = 0; l < NL; ++l) {
        pack_w_kernel<<<(288 * 128 + 255) / 256, blk, 0, stream>>>(msg_W1 + (size_t)l * 288 * 128, msgW1p + (size_t)l * 288 * 128, 288);
        pack_w_kernel<<<(128 * 128 + 255) / 256, blk, 0, stream>>>(msg_W2 + (size_t)l * 128 * 128, msgW2p + (size_t)l * 128 * 128, 128);
        pack_w_kernel<<<(256 * 128 + 255) / 256, blk, 0, stream>>>(upd_W1 + (size_t)l * 256 * 128, updW1p + (size_t)l * 256 * 128, 256);
        pack_w_kernel<<<(128 * 128 + 255) / 256, blk, 0, stream>>>(upd_W2 + (size_t)l * 128 * 128, updW2p + (size_t)l * 128 * 128, 128);
    }
    pack_w_kernel<<<(128 * 128 + 255) / 256, blk, 0, stream>>>(W_out, Woutp, 128);

    node_embed_kernel<<<(N_NODES * NF + 255) / 256, blk, 0, stream>>>(node_coords, W_node, b_node, h, h16);
    edge_embed_kernel<<<(N_EDGES * EF + 255) / 256, blk, 0, stream>>>(edge_dist, W_edge, b_edge, e16);

    const int* src = edge_index;
    const int* dst = edge_index + N_EDGES;
    for (int l = 0; l < NL; ++l) {
        zero_kernel<<<(N_NODES * HD + 255) / 256, blk, 0, stream>>>(agg, N_NODES * HD);
        msg_kernel<<<N_EDGES / 16 / 8, blk, MSG_LDS_BYTES, stream>>>(
            h16, e16, src, dst,
            msgW1p + (size_t)l * 288 * 128, msg_b1 + l * 128,
            msgW2p + (size_t)l * 128 * 128, msg_b2 + l * 128,
            agg, N_EDGES);
        upd_kernel<<<(N_NODES / 16 + 7) / 8, blk, UPD_LDS_BYTES, stream>>>(
            h16, h, agg,
            updW1p + (size_t)l * 256 * 128, upd_b1 + l * 128,
            updW2p + (size_t)l * 128 * 128, upd_b2 + l * 128,
            tmp, N_NODES);
        ln_kernel<<<(N_NODES * 32 + 255) / 256, blk, 0, stream>>>(tmp, ln_g + l * 128, ln_b + l * 128, h, h16, N_NODES);
    }
    out_kernel<<<(N_NODES / 16 + 7) / 8, blk, 0, stream>>>(h16, Woutp, b_out, (float*)d_out, N_NODES);
    (void)in_sizes; (void)n_in; (void)out_size; (void)ws_size;
}
